// GRUModel_40638980555080
// MI455X (gfx1250) — matrix operand order repaired and checked by reading, not compiled, not run on hardware
//
#include <hip/hip_runtime.h>

// ---------------------------------------------------------------------------
// GRU (SEQ=512, BATCH=64, EMB=512, HID=1024) optimized for MI455X (gfx1250).
//  Phase 1: convert W_ih/W_hh to transposed bf16, gather embeddings to bf16.
//  Phase 2: full-chip WMMA bf16 GEMM  gi = X @ W_ih + b_ih  [32768 x 3072].
//  Phase 3: persistent 16-workgroup kernel: per step, gh = h @ W_hh with
//           h staged into LDS via async global->LDS copies (ASYNCcnt,
//           double-buffered 64-col k-chunks overlapping WMMA), GRU
//           elementwise update, double-buffered bf16 h, grid barrier.
//  Phase 4: tiny FC head.
// ---------------------------------------------------------------------------

typedef __attribute__((ext_vector_type(16))) __bf16 v16bf;
typedef __attribute__((ext_vector_type(8)))  __bf16 v8bf;
typedef __attribute__((ext_vector_type(8)))  float  v8f;

#define SEQ     512
#define BATCH   64
#define EMB     512
#define HID     1024
#define OUT_DIM 5
#define W3H     3072      // 3 * HID
#define NWG     16        // workgroups in the sequential phase
#define KCHUNK  64        // k-columns of h staged per LDS chunk
#define NCHUNK  (HID / KCHUNK)          // 16
#define CP_JOBS (BATCH * KCHUNK / 8)    // 512 x b128 copies per chunk

// round-to-nearest-even float -> bf16 (pure bit ops, no libcall)
__device__ __forceinline__ __bf16 f2bf(float f) {
  unsigned u = __builtin_bit_cast(unsigned, f);
  u += 0x7FFFu + ((u >> 16) & 1u);
  unsigned short s = (unsigned short)(u >> 16);
  return __builtin_bit_cast(__bf16, s);
}

// A-operand tile (16x32, M x K), row-major source, base at (m0,k0).
// Lane L (wave32): row = m0 + (L&15); elems 0..7  = K + (L>>4)*8 + 0..7,
//                                      elems 8..15 = K + 16 + (L>>4)*8 + 0..7
__device__ __forceinline__ v16bf ld_a_tile(const __bf16* base, int ld) {
  const int lane = threadIdx.x & 31;
  const __bf16* p = base + (lane & 15) * ld + ((lane >> 4) << 3);
  v8bf lo = *(const v8bf*)(p);
  v8bf hi = *(const v8bf*)(p + 16);
  return __builtin_shufflevector(lo, hi, 0,1,2,3,4,5,6,7,8,9,10,11,12,13,14,15);
}

// B-operand tile (32x16, K x N) loaded from a TRANSPOSED weight matrix
// Wt[N][K] (row-major), base at (n0,k0).  Lane L: col = n0 + (L&15),
// elems 0..7 = K = k0 + (L>>4)*8 + 0..7, elems 8..15 = K = k0 + 16 + (L>>4)*8 + 0..7.
__device__ __forceinline__ v16bf ld_b_tile(const __bf16* base, int ld) {
  const int lane = threadIdx.x & 31;
  const __bf16* p = base + (lane & 15) * ld + ((lane >> 4) << 3);
  v8bf lo = *(const v8bf*)(p);
  v8bf hi = *(const v8bf*)(p + 16);
  return __builtin_shufflevector(lo, hi, 0,1,2,3,4,5,6,7,8,9,10,11,12,13,14,15);
}

__device__ __forceinline__ v8f bfwmma(v16bf a, v16bf b, v8f c) {
  return __builtin_amdgcn_wmma_f32_16x16x32_bf16(false, a, false, b,
                                                 (short)0, c, false, false);
}

// ---------------------------------------------------------------------------
__global__ void k_init(float* __restrict__ h32, __bf16* __restrict__ hb0,
                       unsigned* __restrict__ ctr) {
  long idx = (long)blockIdx.x * blockDim.x + threadIdx.x;
  if (idx == 0) *ctr = 0u;
  if (idx < (long)BATCH * HID) {
    h32[idx] = 0.0f;
    hb0[idx] = f2bf(0.0f);
  }
}

// src [K][N] (row-major f32) -> dst [N][K] (row-major bf16)
__global__ void k_transpose_bf16(const float* __restrict__ src,
                                 __bf16* __restrict__ dst, int K, int N) {
  long idx = (long)blockIdx.x * blockDim.x + threadIdx.x;
  long total = (long)K * N;
  if (idx >= total) return;
  int n = (int)(idx / K);
  int k = (int)(idx % K);
  dst[idx] = f2bf(src[(long)k * N + n]);
}

// X[t*64+b][e] = bf16(emb[text[t][b]][e])
__global__ void k_embed(const int* __restrict__ text,
                        const float* __restrict__ emb,
                        __bf16* __restrict__ X) {
  int row = blockIdx.x;                 // 0..SEQ*BATCH-1
  int tok = text[row];
  const float* e = emb + (long)tok * EMB;
  __bf16* o = X + (long)row * EMB;
  for (int i = threadIdx.x; i < EMB; i += blockDim.x) o[i] = f2bf(e[i]);
}

// gi = X @ W_ih + b_ih ; X [32768 x 512] bf16, Wt_ih [3072 x 512] bf16 (transposed)
// grid (M/32, N/128), 256 threads = 8 waves; wave w: n0 = blk_n*128 + w*16,
// two M tiles (m_base, m_base+16).  32 WMMAs per wave.
__global__ void __launch_bounds__(256)
k_gi_gemm(const __bf16* __restrict__ X, const __bf16* __restrict__ Wt,
          const float* __restrict__ b_ih, float* __restrict__ gi) {
  const int lane   = threadIdx.x & 31;
  const int wave   = threadIdx.x >> 5;
  const int m_base = blockIdx.x * 32;
  const int n0     = blockIdx.y * 128 + wave * 16;

  v8f acc0 = {}; v8f acc1 = {};
  const __bf16* a0 = X + (long)m_base * EMB;
  const __bf16* a1 = X + (long)(m_base + 16) * EMB;
  const __bf16* bp = Wt + (long)n0 * EMB;

  for (int k = 0; k < EMB; k += 32) {
    v16bf b  = ld_b_tile(bp + k, EMB);
    v16bf aa = ld_a_tile(a0 + k, EMB);
    acc0 = bfwmma(aa, b, acc0);
    v16bf ab = ld_a_tile(a1 + k, EMB);
    acc1 = bfwmma(ab, b, acc1);
  }

  const int   col   = n0 + (lane & 15);
  const float bias  = b_ih[col];
  const int   rbase = (lane >> 4) << 3;
#pragma unroll
  for (int r = 0; r < 8; ++r) {
    gi[(long)(m_base + rbase + r) * W3H + col]      = acc0[r] + bias;
    gi[(long)(m_base + 16 + rbase + r) * W3H + col] = acc1[r] + bias;
  }
}

// ---------------------------------------------------------------------------
// Persistent sequential GRU.  NWG=16 blocks x 384 threads (12 waves).
// Block `blk` owns HID columns [blk*64, blk*64+64).  Wave = gate*4 + csub:
// computes gh columns nw..nw+15 of its gate for all 64 batch rows.
// h is staged through LDS in double-buffered 64-col k-chunks via
// global_load_async_to_lds_b128 (ASYNCcnt), overlapping the WMMAs.
// ---------------------------------------------------------------------------

// copy one [64 x 64] bf16 chunk (k-cols k0..k0+63) of h into LDS (async)
__device__ __forceinline__ void issue_chunk_copy(const __bf16* __restrict__ hcur,
                                                 int k0, __bf16* hs_dst) {
  for (int u = threadIdx.x; u < CP_JOBS; u += 384) {   // 512 x 16B, wave-uniform trips
    int row = u >> 3;
    int seg = u & 7;
    const __bf16* g = hcur + row * HID + k0 + seg * 8;
    unsigned lds = (unsigned)(size_t)(hs_dst + row * KCHUNK + seg * 8);
    asm volatile("global_load_async_to_lds_b128 %0, %1, off"
                 :: "v"(lds), "v"(g) : "memory");
  }
}

__global__ void __launch_bounds__(384)
k_gru_seq(const __bf16* __restrict__ Wt_hh, const float* __restrict__ b_hh,
          const float* __restrict__ gi, float* __restrict__ h32,
          __bf16* __restrict__ hbuf0, __bf16* __restrict__ hbuf1,
          unsigned* __restrict__ ctr) {
  __shared__ float  gh_s[3 * BATCH * 64];          // 48 KB
  __shared__ __bf16 hs[2][BATCH * KCHUNK];         // 2 x 8 KB ping-pong

  const int lane = threadIdx.x & 31;
  const int wave = threadIdx.x >> 5;              // 0..11
  const int gate = wave >> 2;                     // 0..2
  const int csub = wave & 3;                      // 0..3
  const int blk  = blockIdx.x;                    // 0..15

  const int nw = gate * HID + blk * 64 + csub * 16;   // weight column base
  const __bf16* bp   = Wt_hh + (long)nw * HID;        // Wt_hh [3072 x 1024]
  const float   bias = b_hh[nw + (lane & 15)];

  __bf16* hb[2] = { hbuf0, hbuf1 };

  for (int t = 0; t < SEQ; ++t) {
    const __bf16* hcur = hb[t & 1];

    // ---- gh tile GEMM: [64 x 16] slice, K = HID, A staged via LDS ----
    v8f acc0 = {}, acc1 = {}, acc2 = {}, acc3 = {};
    issue_chunk_copy(hcur, 0, &hs[0][0]);
    for (int c = 0; c < NCHUNK; ++c) {
      if (c + 1 < NCHUNK) {
        // prefetch next chunk into the other buffer (safe: its last readers
        // finished at the trailing barrier of iteration c-1)
        issue_chunk_copy(hcur, (c + 1) * KCHUNK, &hs[(c + 1) & 1][0]);
        // async loads complete in order: once <=1 newest op remains
        // outstanding, all chunk-c copies from this wave have landed
        asm volatile("s_wait_asynccnt 0x1" ::: "memory");
      } else {
        asm volatile("s_wait_asynccnt 0x0" ::: "memory");
      }
      __syncthreads();                       // all waves' chunk-c data in LDS
      const __bf16* hsc = &hs[c & 1][0];
#pragma unroll
      for (int kk = 0; kk < KCHUNK; kk += 32) {
        v16bf b  = ld_b_tile(bp + c * KCHUNK + kk, HID);
        v16bf a0 = ld_a_tile(hsc + 0 * 16 * KCHUNK + kk, KCHUNK);
        acc0 = bfwmma(a0, b, acc0);
        v16bf a1 = ld_a_tile(hsc + 1 * 16 * KCHUNK + kk, KCHUNK);
        acc1 = bfwmma(a1, b, acc1);
        v16bf a2 = ld_a_tile(hsc + 2 * 16 * KCHUNK + kk, KCHUNK);
        acc2 = bfwmma(a2, b, acc2);
        v16bf a3 = ld_a_tile(hsc + 3 * 16 * KCHUNK + kk, KCHUNK);
        acc3 = bfwmma(a3, b, acc3);
      }
      __syncthreads();     // everyone done reading hs[c&1] before it is reused
    }

    // ---- stage gh (+ b_hh) into LDS: gh_s[gate][row][64] ----
    {
      const int colL  = csub * 16 + (lane & 15);
      const int rbase = (lane >> 4) << 3;
      float* gs = gh_s + gate * (BATCH * 64);
#pragma unroll
      for (int r = 0; r < 8; ++r) {
        gs[(0  + rbase + r) * 64 + colL] = acc0[r] + bias;
        gs[(16 + rbase + r) * 64 + colL] = acc1[r] + bias;
        gs[(32 + rbase + r) * 64 + colL] = acc2[r] + bias;
        gs[(48 + rbase + r) * 64 + colL] = acc3[r] + bias;
      }
    }
    __syncthreads();

    // ---- elementwise GRU update for owned columns ----
    __bf16* hnext = hb[(t + 1) & 1];
    const float* git = gi + (long)t * BATCH * W3H;
    for (int i = threadIdx.x; i < BATCH * 64; i += blockDim.x) {
      int b   = i >> 6;
      int j   = i & 63;
      int col = blk * 64 + j;
      const float* g = git + (long)b * W3H;
      float rr = 1.0f / (1.0f + __expf(-(g[col]           + gh_s[0 * 4096 + b * 64 + j])));
      float zz = 1.0f / (1.0f + __expf(-(g[HID + col]     + gh_s[1 * 4096 + b * 64 + j])));
      float nn = tanhf(  g[2 * HID + col] + rr *            gh_s[2 * 4096 + b * 64 + j]);
      float hp = h32[b * HID + col];
      float hv = (1.0f - zz) * nn + zz * hp;
      h32[b * HID + col]   = hv;
      hnext[b * HID + col] = f2bf(hv);
    }

    // prefetch next step's gi slice into cache (global_prefetch_b8)
    if (t + 1 < SEQ) {
      const float* pf = gi + (long)(t + 1) * BATCH * W3H + (long)blk * 64;
      __builtin_prefetch(pf + (size_t)threadIdx.x * 64, 0, 1);
    }

    // ---- grid-wide barrier (monotonic counter, one bump per WG per step) ----
    __threadfence();
    __syncthreads();
    if (threadIdx.x == 0) {
      __hip_atomic_fetch_add(ctr, 1u, __ATOMIC_RELEASE, __HIP_MEMORY_SCOPE_AGENT);
      const unsigned target = (unsigned)(NWG * (t + 1));
      while (__hip_atomic_load(ctr, __ATOMIC_ACQUIRE, __HIP_MEMORY_SCOPE_AGENT) < target) {
        __builtin_amdgcn_s_sleep(2);
      }
    }
    __syncthreads();
    __threadfence();
  }
}

// out[b][o] = h[b] . fc_W[:,o] + fc_b[o]    (64x5, trivial)
__global__ void k_fc(const float* __restrict__ h32, const float* __restrict__ fcW,
                     const float* __restrict__ fcb, float* __restrict__ out) {
  int idx = blockIdx.x * blockDim.x + threadIdx.x;
  if (idx >= BATCH * OUT_DIM) return;
  int b = idx / OUT_DIM, o = idx % OUT_DIM;
  float s = fcb[o];
  for (int k = 0; k < HID; ++k) s += h32[b * HID + k] * fcW[k * OUT_DIM + o];
  out[idx] = s;
}

// ---------------------------------------------------------------------------
extern "C" void kernel_launch(void* const* d_in, const int* in_sizes, int n_in,
                              void* d_out, int out_size, void* d_ws, size_t ws_size,
                              hipStream_t stream) {
  const int*   text = (const int*)  d_in[0];
  const float* emb  = (const float*)d_in[1];
  const float* W_ih = (const float*)d_in[2];
  const float* W_hh = (const float*)d_in[3];
  const float* b_ih = (const float*)d_in[4];
  const float* b_hh = (const float*)d_in[5];
  const float* fcW  = (const float*)d_in[6];
  const float* fcb  = (const float*)d_in[7];
  float* out = (float*)d_out;
  (void)in_sizes; (void)n_in; (void)out_size; (void)ws_size;

  // workspace carve-up (256B aligned)
  char*  ws  = (char*)d_ws;
  size_t off = 0;
  auto carve = [&](size_t bytes) -> void* {
    void* p = ws + off;
    off += (bytes + 255) & ~(size_t)255;
    return p;
  };
  unsigned* ctr   = (unsigned*)carve(256);
  __bf16*   Wt_ih = (__bf16*)  carve((size_t)W3H * EMB * sizeof(short));   //  3 MB
  __bf16*   Wt_hh = (__bf16*)  carve((size_t)W3H * HID * sizeof(short));   //  6 MB
  __bf16*   X     = (__bf16*)  carve((size_t)SEQ * BATCH * EMB * sizeof(short)); // 32 MB
  float*    h32   = (float*)   carve((size_t)BATCH * HID * sizeof(float));
  __bf16*   hb0   = (__bf16*)  carve((size_t)BATCH * HID * sizeof(short));
  __bf16*   hb1   = (__bf16*)  carve((size_t)BATCH * HID * sizeof(short));
  float*    gi    = (float*)   carve((size_t)SEQ * BATCH * W3H * sizeof(float)); // 384 MB

  // 1) zero h / barrier counter
  k_init<<<dim3(256), dim3(256), 0, stream>>>(h32, hb0, ctr);

  // 2) weight transposes -> bf16
  {
    long tot = (long)W3H * EMB;
    k_transpose_bf16<<<dim3((tot + 255) / 256), dim3(256), 0, stream>>>(W_ih, Wt_ih, EMB, W3H);
  }
  {
    long tot = (long)W3H * HID;
    k_transpose_bf16<<<dim3((tot + 255) / 256), dim3(256), 0, stream>>>(W_hh, Wt_hh, HID, W3H);
  }

  // 3) embedding gather -> bf16
  k_embed<<<dim3(SEQ * BATCH), dim3(128), 0, stream>>>(text, emb, X);

  // 4) full-chip WMMA GEMM: gi = X @ W_ih + b_ih
  k_gi_gemm<<<dim3(SEQ * BATCH / 32, W3H / 128), dim3(256), 0, stream>>>(X, Wt_ih, b_ih, gi);

  // 5) persistent sequential GRU with grid barrier + async LDS staging
  k_gru_seq<<<dim3(NWG), dim3(384), 0, stream>>>(Wt_hh, b_hh, gi, h32, hb0, hb1, ctr);

  // 6) FC head
  k_fc<<<dim3((BATCH * OUT_DIM + 63) / 64), dim3(64), 0, stream>>>(h32, fcW, fcb, out);
}
